// SparseEncoder4D_12764642804259
// MI455X (gfx1250) — matrix operand order repaired and checked by reading, not compiled, not run on hardware
//
#include <hip/hip_runtime.h>
#include <hip/hip_bf16.h>

typedef __attribute__((ext_vector_type(16))) _Float16 v16h;
typedef __attribute__((ext_vector_type(8)))  _Float16 v8h;
typedef __attribute__((ext_vector_type(8)))  float    v8f;
typedef __attribute__((ext_vector_type(4)))  float    v4f;

#define KOFF   81          // kernel offsets
#define CHID   8           // hidden channels
#define NGRP   21          // ceil(81/4) groups of 4 offsets per WMMA
#define DIM_T  4
#define DIM_Z  32
#define DIM_Y  256
#define DIM_X  256

// ---------------- kernel 1: zero the dense output (64 MB) ----------------
__global__ __launch_bounds__(256) void zero_f4(v4f* __restrict__ p, int n4) {
    int i = blockIdx.x * blockDim.x + threadIdx.x;
    int stride = gridDim.x * blockDim.x;
    v4f z = {0.f, 0.f, 0.f, 0.f};
    for (; i < n4; i += stride) __builtin_nontemporal_store(z, p + i);
}

// ---------------- kernel 2: conv1 (1 -> 8), f32 accumulate, f16 out ----------------
__global__ __launch_bounds__(256) void conv1_kernel(
    const float* __restrict__ feats, const float* __restrict__ w1,
    const int* __restrict__ nbr_idx, const int* __restrict__ nbr_mask,
    _Float16* __restrict__ h1h, int N)
{
    __shared__ float w1s[KOFF * CHID];
    for (int i = threadIdx.x; i < KOFF * CHID; i += blockDim.x) w1s[i] = w1[i];
    __syncthreads();

    int n = blockIdx.x * blockDim.x + threadIdx.x;
    if (n >= N) return;

    float acc[CHID];
#pragma unroll
    for (int d = 0; d < CHID; ++d) acc[d] = 0.f;

    const int base = n * KOFF;
    for (int k = 0; k < KOFF; ++k) {
        // single-use streams: non-temporal; branchless gather (feats table is L2-hot)
        int m  = __builtin_nontemporal_load(nbr_mask + base + k);
        int id = __builtin_nontemporal_load(nbr_idx + base + k);
        float f = feats[id];
        float s = m ? f : 0.f;
#pragma unroll
        for (int d = 0; d < CHID; ++d) acc[d] = fmaf(s, w1s[k * CHID + d], acc[d]);
    }
    v8h hv;
#pragma unroll
    for (int d = 0; d < CHID; ++d) hv[d] = (_Float16)fmaxf(acc[d], 0.f);
    *(v8h*)(h1h + (size_t)n * CHID) = hv;   // 16B row, 16B aligned
}

// ---------------- kernel 3: build WMMA B-fragments for w2 ----------------
// Layout per ISA (16-bit B 32x16, wave32): lanes 0-15 hold K=0..7,16..23, lanes 16-31
// hold K=8..15,24..31, column N = lane&15.  K = 8*i + c maps to (offset 4g+i, chan c).
// Invalid offsets (k >= 81) and columns n >= 8 are built as ZERO: this lets the
// GEMM kernel feed garbage A rows into those K-slots with no effect.
__global__ __launch_bounds__(256) void build_fragB(
    const float* __restrict__ w2, _Float16* __restrict__ fragB)
{
    int t = blockIdx.x * blockDim.x + threadIdx.x;
    if (t >= NGRP * 32) return;
    int g = t >> 5, lane = t & 31;
    int half = lane >> 4, n = lane & 15;
    v16h vals;
#pragma unroll
    for (int j = 0; j < 16; ++j) {
        int K = (j < 8) ? (8 * half + j) : (16 + 8 * half + (j - 8));
        int i = K >> 3, c = K & 7;
        int k = 4 * g + i;
        float w = (n < CHID && k < KOFF) ? w2[(k * CHID + c) * CHID + n] : 0.f;
        vals[j] = (_Float16)w;
    }
    *(v16h*)(fragB + (size_t)t * 16) = vals;  // 32B per lane, 32B aligned
}

// ---------------- kernel 4: conv2 (8->8) via WMMA + head + dense scatter ----------------
// One wave per 16-voxel tile.  A-fragment (16-bit A 16x32 layout): lane L holds
// row M=L&15; low 8 halves = K 0..7 (lanes<16) / 8..15 (lanes>=16), high 8 =
// K 16..23 / 24..31  =>  offsets i_lo = lane>>4, i_hi = i_lo + 2, channels 0..7.
union AFrag { v16h v; struct { v8h lo; v8h hi; } s; };

__device__ __forceinline__ v8h gather_row(
    const _Float16* __restrict__ h1h, const int* __restrict__ nbr_idx,
    const int* __restrict__ nbr_mask, int base, int k)
{
    // idx+mask issue in parallel (both non-temporal); gather is unconditional
    // (h1 table is L2-resident); masked rows are zeroed with vector selects.
    int m  = __builtin_nontemporal_load(nbr_mask + base + k);
    int id = __builtin_nontemporal_load(nbr_idx + base + k);
    v8h row = *(const v8h*)(h1h + (size_t)id * CHID);
    v8h z = {};
    return m ? row : z;
}

__global__ __launch_bounds__(256) void conv2_head_scatter(
    const _Float16* __restrict__ h1h, const _Float16* __restrict__ fragB,
    const int* __restrict__ nbr_idx, const int* __restrict__ nbr_mask,
    const float* __restrict__ w_out, const int* __restrict__ coords,
    const int* __restrict__ batch_idx, float* __restrict__ dense, int ntiles)
{
    int lane = threadIdx.x & 31;
    int wave = threadIdx.x >> 5;
    int tile = blockIdx.x * (blockDim.x >> 5) + wave;
    if (tile >= ntiles) return;            // wave-uniform: EXEC stays full below

    const int row  = lane & 15;            // voxel row within tile
    const int half = lane >> 4;            // 0 or 1
    const int n    = tile * 16 + row;      // N is a multiple of 16 -> always valid
    const int base = n * KOFF;

    // two independent accumulator chains to hide WMMA latency
    v8f acc0 = {}, acc1 = {};
    AFrag a0, a1;
    // main loop: groups 0..19 in pairs; all k in range (max k = 4*19+3 = 79)
    for (int g = 0; g < NGRP - 1; g += 2) {
        int k0 = 4 * g + half;
        a0.s.lo = gather_row(h1h, nbr_idx, nbr_mask, base, k0);
        a0.s.hi = gather_row(h1h, nbr_idx, nbr_mask, base, k0 + 2);
        a1.s.lo = gather_row(h1h, nbr_idx, nbr_mask, base, k0 + 4);
        a1.s.hi = gather_row(h1h, nbr_idx, nbr_mask, base, k0 + 6);
        v16h b0 = *(const v16h*)(fragB + ((size_t)(g * 32 + lane) << 4));
        v16h b1 = *(const v16h*)(fragB + ((size_t)((g + 1) * 32 + lane) << 4));
        acc0 = __builtin_amdgcn_wmma_f32_16x16x32_f16(
            false, a0.v, false, b0, (short)0, acc0, false, false);
        acc1 = __builtin_amdgcn_wmma_f32_16x16x32_f16(
            false, a1.v, false, b1, (short)0, acc1, false, false);
    }
    // tail: group 20 covers only k=80; clamp OOB k to 80 -- the corresponding
    // B rows are zero, so the garbage A data contributes nothing.
    {
        int k0 = 4 * (NGRP - 1) + half;            // 80 or 81
        int k1 = k0 + 2;                           // 82 or 83
        k0 = k0 > (KOFF - 1) ? (KOFF - 1) : k0;
        k1 = k1 > (KOFF - 1) ? (KOFF - 1) : k1;
        a0.s.lo = gather_row(h1h, nbr_idx, nbr_mask, base, k0);
        a0.s.hi = gather_row(h1h, nbr_idx, nbr_mask, base, k1);
        v16h b0 = *(const v16h*)(fragB + ((size_t)((NGRP - 1) * 32 + lane) << 4));
        acc0 = __builtin_amdgcn_wmma_f32_16x16x32_f16(
            false, a0.v, false, b0, (short)0, acc0, false, false);
    }
    v8f acc = acc0 + acc1;

    // D layout: lane holds column (= out channel) ch = lane&15, rows M = 8*half + r.
    // head: out[m] = sum_c relu(h2[m,c]) * w_out[c]  -> xor-reduce over 8 channel lanes.
    const int ch = lane & 15;
    float wo = (ch < CHID) ? w_out[ch] : 0.f;
    float v[8];
#pragma unroll
    for (int r = 0; r < 8; ++r) v[r] = fmaxf(acc[r], 0.f) * wo;
#pragma unroll
    for (int m = 1; m <= 4; m <<= 1) {
#pragma unroll
        for (int r = 0; r < 8; ++r) v[r] += __shfl_xor(v[r], m, 32);
    }

    if ((lane & 15) == 0) {                // lanes 0 (rows 0-7) and 16 (rows 8-15)
#pragma unroll
        for (int r = 0; r < 8; ++r) {
            int nv = tile * 16 + 8 * half + r;
            const int4 c4 = *(const int4*)(coords + (size_t)nv * 4); // x,y,z,t
            int b = batch_idx[nv];
            long idx = ((((long)b * DIM_T + c4.w) * DIM_Z + c4.z) * DIM_Y + c4.y)
                         * DIM_X + c4.x;
            dense[idx] = v[r];
        }
    }
}

extern "C" void kernel_launch(void* const* d_in, const int* in_sizes, int n_in,
                              void* d_out, int out_size, void* d_ws, size_t ws_size,
                              hipStream_t stream) {
    const float* feats     = (const float*)d_in[0];
    const float* w1        = (const float*)d_in[1];
    const float* w2        = (const float*)d_in[2];
    const float* w_out     = (const float*)d_in[3];
    const int*   nbr_idx   = (const int*)d_in[4];
    const int*   nbr_mask  = (const int*)d_in[5];
    const int*   coords    = (const int*)d_in[6];
    const int*   batch_idx = (const int*)d_in[7];
    float* dense = (float*)d_out;

    const int N = in_sizes[0];             // 150000 voxels (multiple of 16)

    // scratch layout: [0, 21.5KB) B-fragments | [32KB, 32KB + N*16B) f16 h1
    _Float16* fragB = (_Float16*)d_ws;
    _Float16* h1h   = (_Float16*)((char*)d_ws + 32768);

    // 1) zero dense output
    int n4 = out_size >> 2;
    zero_f4<<<(n4 + 255) / 256, 256, 0, stream>>>((v4f*)dense, n4);

    // 2) conv1: 1 -> 8, f16 table for WMMA gathers
    conv1_kernel<<<(N + 255) / 256, 256, 0, stream>>>(
        feats, w1, nbr_idx, nbr_mask, h1h, N);

    // 3) precompute B-fragments (21 groups x 32 lanes x 32B)
    build_fragB<<<(NGRP * 32 + 255) / 256, 256, 0, stream>>>(w2, fragB);

    // 4) conv2 via v_wmma_f32_16x16x32_f16 + head + scatter (1 wave / 16 voxels)
    int ntiles = N / 16;
    int waves_per_block = 256 / 32;
    conv2_head_scatter<<<(ntiles + waves_per_block - 1) / waves_per_block, 256, 0, stream>>>(
        h1h, fragB, nbr_idx, nbr_mask, w_out, coords, batch_idx, dense, ntiles);
}
